// LAS_6579889898185
// MI455X (gfx1250) — matrix operand order repaired and checked by reading, not compiled, not run on hardware
//
#include <hip/hip_runtime.h>

typedef __attribute__((ext_vector_type(16))) _Float16 v16h;
typedef __attribute__((ext_vector_type(8)))  _Float16 v8h;
typedef __attribute__((ext_vector_type(8)))  float    v8f;

#define B_   32
#define T_   500
#define L_   100
#define H_   256
#define NCLS 29
#define ROWS (B_ * T_)   // 16000

__device__ inline float sigm(float x) { return 1.f / (1.f + __expf(-x)); }
__device__ inline float tanh_fast(float x) {
    x = fminf(fmaxf(x, -15.f), 15.f);
    float e = __expf(2.f * x);
    return (e - 1.f) / (e + 1.f);
}
__device__ inline float wred_max(float v) {
    #pragma unroll
    for (int o = 16; o > 0; o >>= 1) v = fmaxf(v, __shfl_xor(v, o, 32));
    return v;
}
__device__ inline float wred_sum(float v) {
    #pragma unroll
    for (int o = 16; o > 0; o >>= 1) v += __shfl_xor(v, o, 32);
    return v;
}

__device__ inline v8f wmma_f16(v16h a, v16h b, v8f c) {
    return __builtin_amdgcn_wmma_f32_16x16x32_f16(false, a, false, b, (short)0, c, false, false);
}

// Opaque zero: defeats LICM on loop-invariant weight-fragment loads (hoisting
// 32+ v16h fragments spills to scratch) WITHOUT destroying the pointer's
// global address-space provenance, so loads stay global_load_b128 (LOADcnt
// only, no flat/DScnt coupling).
__device__ __forceinline__ int launder_zero() {
    int z = 0;
    asm volatile("" : "+v"(z));
    return z;
}

// A-fragment (16x32 f16): lane = row (lane&15); lanes 0-15 hold
// K {0..15}, lanes 16-31 hold K {16..31} within a k0 chunk (matches B loads).
__device__ inline v16h load_afrag(const _Float16* rowptr, int k0, int sel) {
    v8h lo = *(const v8h*)(rowptr + k0 + sel * 16);
    v8h hi = *(const v8h*)(rowptr + k0 + sel * 16 + 8);
    v16h a;
    #pragma unroll
    for (int i = 0; i < 8; ++i) { a[i] = lo[i]; a[i + 8] = hi[i]; }
    return a;
}

// ---------------- conversion kernels ----------------
__global__ void k_cvt_pad(const float* __restrict__ src, _Float16* __restrict__ dst,
                          int rows, int kin, int kout) {
    long n = (long)rows * kout;
    for (long i = (long)blockIdx.x * blockDim.x + threadIdx.x; i < n;
         i += (long)gridDim.x * blockDim.x) {
        long r = i / kout; int c = (int)(i - r * kout);
        dst[i] = (c < kin) ? (_Float16)src[r * kin + c] : (_Float16)0.f;
    }
}

// Wd[j][c] : c<512 -> dec_Wih[j][29+c] ; else dec_Whh[j][c-512]
__global__ void k_make_decW(const float* __restrict__ Wih, const float* __restrict__ Whh,
                            _Float16* __restrict__ Wd) {
    long n = 1024L * 768;
    for (long i = (long)blockIdx.x * blockDim.x + threadIdx.x; i < n;
         i += (long)gridDim.x * blockDim.x) {
        long j = i / 768; int c = (int)(i - j * 768);
        float v = (c < 512) ? Wih[j * 541 + 29 + c] : Whh[j * 256 + (c - 512)];
        Wd[i] = (_Float16)v;
    }
}
__global__ void k_make_tokW(const float* __restrict__ Wih, _Float16* __restrict__ tokW) {
    int n = NCLS * 1024;
    for (int i = blockIdx.x * blockDim.x + threadIdx.x; i < n; i += gridDim.x * blockDim.x) {
        int t = i / 1024, j = i - t * 1024;
        tokW[i] = (_Float16)Wih[(long)j * 541 + t];
    }
}

// ---------------- WMMA GEMM: C[M][ldc] = A[M][lda] @ W[N][ldw]^T (+bias) ----
// Each wave computes a 16 x (16*NT) strip: one A-fragment load feeds NT WMMAs.
// grid.x = M/16, grid.y = N/(128*NT), 8 waves/block.
template <int NT>
__global__ void __launch_bounds__(256)
k_gemm16(const _Float16* __restrict__ A, int lda,
         const _Float16* __restrict__ W, int ldw,
         const float* __restrict__ bias,
         _Float16* __restrict__ C, int ldc, int kiters) {
    const int wave = threadIdx.x >> 5, lane = threadIdx.x & 31;
    const int m0 = blockIdx.x * 16;
    const int n0 = (blockIdx.y * 8 + wave) * (16 * NT);
    const int colL = lane & 15, sel = lane >> 4;

    v8f zero = {};
    v8f acc[NT];
    #pragma unroll
    for (int j = 0; j < NT; ++j) acc[j] = zero;

    const _Float16* ar = A + (size_t)(m0 + colL) * lda;
    const _Float16* wr[NT];
    #pragma unroll
    for (int j = 0; j < NT; ++j) wr[j] = W + (size_t)(n0 + 16 * j + colL) * ldw;

    for (int ki = 0; ki < kiters; ++ki) {
        int k0 = ki * 32;
        v16h a = load_afrag(ar, k0, sel);
        #pragma unroll
        for (int j = 0; j < NT; ++j)
            acc[j] = wmma_f16(a, *(const v16h*)(wr[j] + k0 + sel * 16), acc[j]);
    }
    #pragma unroll
    for (int j = 0; j < NT; ++j) {
        float bv = bias ? bias[n0 + 16 * j + colL] : 0.f;
        #pragma unroll
        for (int v = 0; v < 8; ++v)
            C[(size_t)(m0 + v + sel * 8) * ldc + n0 + 16 * j + colL] =
                (_Float16)(acc[j][v] + bv);
    }
}

// ---------------- LSTM scan (one direction per block) ----------------------
// xz (f16, bias included) precomputed [B*T][1024]; h kept f16 in LDS; c in regs.
#define HSLD 272
__global__ void __launch_bounds__(1024)
k_lstm_scan(const _Float16* __restrict__ xzf, const _Float16* __restrict__ xzb,
            const _Float16* __restrict__ Whhf, const _Float16* __restrict__ Whhb,
            _Float16* __restrict__ out /* [B*T][512] */) {
    const int dir = blockIdx.x;
    const _Float16* xz  = dir ? xzb  : xzf;
    const _Float16* Whh = dir ? Whhb : Whhf;

    __shared__ _Float16 hs[B_ * HSLD];

    const int tid = threadIdx.x, wave = tid >> 5, lane = tid & 31;
    const int m0 = (wave & 1) * 16;
    const int n0 = (wave >> 1) * 16;
    const int colL = lane & 15, sel = lane >> 4;
    const int col = n0 + colL;

    for (int i = tid; i < B_ * HSLD; i += 1024) hs[i] = (_Float16)0.f;
    __syncthreads();

    float creg[8];
    #pragma unroll
    for (int v = 0; v < 8; ++v) creg[v] = 0.f;

    const _Float16* bI = Whh + (size_t)(0 * H_ + col) * H_;
    const _Float16* bF = Whh + (size_t)(1 * H_ + col) * H_;
    const _Float16* bG = Whh + (size_t)(2 * H_ + col) * H_;
    const _Float16* bO = Whh + (size_t)(3 * H_ + col) * H_;
    const _Float16* ar = hs + (m0 + colL) * HSLD;

    // warm the recurrent weight rows into cache (re-read 500x)
    __builtin_prefetch(bI, 0, 3);
    __builtin_prefetch(bF, 0, 3);
    __builtin_prefetch(bG, 0, 3);
    __builtin_prefetch(bO, 0, 3);

    for (int s = 0; s < T_; ++s) {
        const int t = dir ? (T_ - 1 - s) : s;
        // opaque zero offset: keep weight loads inside the loop as global loads
        const int zo = launder_zero();
        const _Float16* pI = bI + zo;
        const _Float16* pF = bF + zo;
        const _Float16* pG = bG + zo;
        const _Float16* pO = bO + zo;
        v8f aI = {}, aF = {}, aG = {}, aO = {};
        #pragma unroll
        for (int k0 = 0; k0 < H_; k0 += 32) {
            v16h a = load_afrag(ar, k0, sel);
            aI = wmma_f16(a, *(const v16h*)(pI + k0 + sel * 16), aI);
            aF = wmma_f16(a, *(const v16h*)(pF + k0 + sel * 16), aF);
            aG = wmma_f16(a, *(const v16h*)(pG + k0 + sel * 16), aG);
            aO = wmma_f16(a, *(const v16h*)(pO + k0 + sel * 16), aO);
        }
        float hv[8];
        #pragma unroll
        for (int v = 0; v < 8; ++v) {
            int row = m0 + v + sel * 8;                // batch index
            const _Float16* xr = xz + ((size_t)row * T_ + t) * 1024;
            float zi = aI[v] + (float)xr[0 * H_ + col];
            float zf = aF[v] + (float)xr[1 * H_ + col];
            float zg = aG[v] + (float)xr[2 * H_ + col];
            float zo2 = aO[v] + (float)xr[3 * H_ + col];
            creg[v] = sigm(zf) * creg[v] + sigm(zi) * tanh_fast(zg);
            hv[v]   = sigm(zo2) * tanh_fast(creg[v]);
        }
        __syncthreads();   // everyone done reading old h
        #pragma unroll
        for (int v = 0; v < 8; ++v) {
            int row = m0 + v + sel * 8;
            _Float16 hh = (_Float16)hv[v];
            hs[row * HSLD + col] = hh;
            out[((size_t)row * T_ + t) * 512 + dir * H_ + col] = hh;
        }
        __syncthreads();
    }
}

// ---------------- Attention decoder: one persistent block --------------------
__global__ void __launch_bounds__(1024)
k_decoder(const _Float16* __restrict__ encproj, // [B*T][256]
          const _Float16* __restrict__ enc,     // [B*T][512]
          const _Float16* __restrict__ W2h16,   // [256][256]
          const _Float16* __restrict__ Wd,      // [1024][768]
          const _Float16* __restrict__ tokW,    // [29][1024]
          const float* __restrict__ dec_b, const float* __restrict__ V,
          const float* __restrict__ Wout, const float* __restrict__ bout,
          const int* __restrict__ targets, float* __restrict__ out) {
    extern __shared__ char smem[];
    float* sc  = (float*)smem;          // [32][500]
    float* w2h = sc + B_ * T_;          // [32][256]
    float* vsm = w2h + B_ * H_;         // [256]
    _Float16* af = (_Float16*)(vsm + H_); // [32][768]: ctx 0..511, h 512..767

    const int tid = threadIdx.x, wave = tid >> 5, lane = tid & 31;
    const int m0 = (wave & 1) * 16;
    const int n0 = (wave >> 1) * 16;
    const int colL = lane & 15, sel = lane >> 4;
    const int col = n0 + colL;

    for (int i = tid; i < B_ * 768; i += 1024) af[i] = (_Float16)0.f;
    for (int i = tid; i < H_; i += 1024) vsm[i] = V[i];
    __syncthreads();

    float creg[8];
    #pragma unroll
    for (int v = 0; v < 8; ++v) creg[v] = 0.f;

    const float dbI = dec_b[0 * H_ + col], dbF = dec_b[1 * H_ + col];
    const float dbG = dec_b[2 * H_ + col], dbO = dec_b[3 * H_ + col];

    for (int l = 0; l < L_; ++l) {
        { // 1: w2h = h @ W2^T   (A = h slice of af)
            v8f acc = {};
            const _Float16* ar = af + (m0 + colL) * 768 + 512;
            const _Float16* br = W2h16 + (size_t)col * H_ + launder_zero();
            #pragma unroll
            for (int k0 = 0; k0 < H_; k0 += 32) {
                v16h a = load_afrag(ar, k0, sel);
                acc = wmma_f16(a, *(const v16h*)(br + k0 + sel * 16), acc);
            }
            #pragma unroll
            for (int v = 0; v < 8; ++v) w2h[(m0 + v + sel * 8) * H_ + col] = acc[v];
        }
        __syncthreads();
        // 2: scores[b][t] = V . tanh(enc_proj + w2h)
        for (int p = tid; p < B_ * T_; p += 1024) {
            int b = p / T_, t = p - b * T_;
            const _Float16* ep = encproj + ((size_t)b * T_ + t) * H_;
            const float* wh = w2h + b * H_;
            float s = 0.f;
            for (int d = 0; d < H_; ++d) s += vsm[d] * tanh_fast((float)ep[d] + wh[d]);
            sc[p] = s;
        }
        __syncthreads();
        { // 3: softmax over T, one wave per batch
            int b = wave;
            float mx = -3.4e38f;
            for (int t = lane; t < T_; t += 32) mx = fmaxf(mx, sc[b * T_ + t]);
            mx = wred_max(mx);
            float sum = 0.f;
            for (int t = lane; t < T_; t += 32) {
                float e = __expf(sc[b * T_ + t] - mx);
                sc[b * T_ + t] = e; sum += e;
            }
            sum = wred_sum(sum);
            float inv = 1.f / sum;
            for (int t = lane; t < T_; t += 32) sc[b * T_ + t] *= inv;
        }
        __syncthreads();
        // 4: ctx[b][e] = sum_t a * enc  -> af[.][0..511]
        for (int q = tid; q < B_ * 512; q += 1024) {
            int b = q >> 9, e = q & 511;
            const _Float16* er = enc + (size_t)b * T_ * 512 + e;
            const float* ab = sc + b * T_;
            float s = 0.f;
            for (int t = 0; t < T_; ++t) s += ab[t] * (float)er[(size_t)t * 512];
            af[b * 768 + e] = (_Float16)s;
        }
        __syncthreads();
        { // 5: z = [ctx|h] @ Wd^T + onehot-col + b ; gates; update c,h
            v8f aI = {}, aF = {}, aG = {}, aO = {};
            const int zo = launder_zero();
            const _Float16* ar = af + (m0 + colL) * 768;
            const _Float16* bi = Wd + (size_t)(0 * H_ + col) * 768 + zo;
            const _Float16* bf = Wd + (size_t)(1 * H_ + col) * 768 + zo;
            const _Float16* bg = Wd + (size_t)(2 * H_ + col) * 768 + zo;
            const _Float16* bo = Wd + (size_t)(3 * H_ + col) * 768 + zo;
            for (int k0 = 0; k0 < 768; k0 += 32) {
                v16h a = load_afrag(ar, k0, sel);
                aI = wmma_f16(a, *(const v16h*)(bi + k0 + sel * 16), aI);
                aF = wmma_f16(a, *(const v16h*)(bf + k0 + sel * 16), aF);
                aG = wmma_f16(a, *(const v16h*)(bg + k0 + sel * 16), aG);
                aO = wmma_f16(a, *(const v16h*)(bo + k0 + sel * 16), aO);
            }
            __syncthreads();  // all waves done reading af(h)
            float hv[8];
            #pragma unroll
            for (int v = 0; v < 8; ++v) {
                int row = m0 + v + sel * 8;  // batch
                int tok = (l == 0) ? 0 : targets[row * L_ + l - 1];
                const _Float16* tw = tokW + tok * 1024;
                float zi = aI[v] + (float)tw[0 * H_ + col] + dbI;
                float zf = aF[v] + (float)tw[1 * H_ + col] + dbF;
                float zg = aG[v] + (float)tw[2 * H_ + col] + dbG;
                float zo2 = aO[v] + (float)tw[3 * H_ + col] + dbO;
                creg[v] = sigm(zf) * creg[v] + sigm(zi) * tanh_fast(zg);
                hv[v]   = sigm(zo2) * tanh_fast(creg[v]);
            }
            #pragma unroll
            for (int v = 0; v < 8; ++v)
                af[(m0 + v + sel * 8) * 768 + 512 + col] = (_Float16)hv[v];
        }
        __syncthreads();
        // 6: out = [h|ctx] @ Wout^T + bout
        for (int q = tid; q < B_ * NCLS; q += 1024) {
            int b = q / NCLS, cls = q - b * NCLS;
            const float* wr = Wout + cls * 768;
            const _Float16* ab = af + b * 768;
            float s = bout[cls];
            for (int d = 0; d < H_; ++d)  s += wr[d] * (float)ab[512 + d];      // h part
            for (int d = 0; d < 512; ++d) s += wr[H_ + d] * (float)ab[d];       // ctx part
            out[((size_t)b * L_ + l) * NCLS + cls] = s;
        }
        __syncthreads();
    }
}

// ---------------- host ----------------
extern "C" void kernel_launch(void* const* d_in, const int* in_sizes, int n_in,
                              void* d_out, int out_size, void* d_ws, size_t ws_size,
                              hipStream_t stream) {
    (void)in_sizes; (void)n_in; (void)out_size; (void)ws_size;
    const float* x       = (const float*)d_in[0];
    const int*   targets = (const int*)d_in[1];
    const float *Wih[3][2], *Whh[3][2], *bE[3][2];
    for (int l = 0; l < 3; ++l) {
        int b = 2 + l * 6;
        Wih[l][0] = (const float*)d_in[b + 0]; Whh[l][0] = (const float*)d_in[b + 1];
        bE[l][0]  = (const float*)d_in[b + 2];
        Wih[l][1] = (const float*)d_in[b + 3]; Whh[l][1] = (const float*)d_in[b + 4];
        bE[l][1]  = (const float*)d_in[b + 5];
    }
    const float* W1     = (const float*)d_in[20];
    const float* W2     = (const float*)d_in[21];
    const float* V      = (const float*)d_in[22];
    const float* dWih   = (const float*)d_in[23];
    const float* dWhh   = (const float*)d_in[24];
    const float* dec_b  = (const float*)d_in[25];
    const float* Wout   = (const float*)d_in[26];
    const float* bout   = (const float*)d_in[27];

    char* base = (char*)d_ws; size_t off = 0;
    auto alloc = [&](size_t bytes) -> void* {
        void* p = base + off; off = (off + bytes + 255) & ~(size_t)255; return p;
    };
    _Float16* xpad  = (_Float16*)alloc((size_t)ROWS * 96 * 2);
    _Float16* actA  = (_Float16*)alloc((size_t)ROWS * 512 * 2);
    _Float16* actB  = (_Float16*)alloc((size_t)ROWS * 512 * 2);
    _Float16* xzf   = (_Float16*)alloc((size_t)ROWS * 1024 * 2);
    _Float16* xzb   = (_Float16*)alloc((size_t)ROWS * 1024 * 2);
    _Float16* eproj = (_Float16*)alloc((size_t)ROWS * 256 * 2);
    _Float16 *WihF[3][2], *WhhF[3][2];
    for (int l = 0; l < 3; ++l) {
        int kp = (l == 0) ? 96 : 512;
        for (int d = 0; d < 2; ++d) {
            WihF[l][d] = (_Float16*)alloc((size_t)1024 * kp * 2);
            WhhF[l][d] = (_Float16*)alloc((size_t)1024 * 256 * 2);
        }
    }
    _Float16* W1F  = (_Float16*)alloc((size_t)256 * 512 * 2);
    _Float16* W2F  = (_Float16*)alloc((size_t)256 * 256 * 2);
    _Float16* WdF  = (_Float16*)alloc((size_t)1024 * 768 * 2);
    _Float16* tokW = (_Float16*)alloc((size_t)NCLS * 1024 * 2);

    const int CG = 2048;
    // conversions / repacks
    k_cvt_pad<<<CG, 256, 0, stream>>>(x, xpad, ROWS, 80, 96);
    for (int l = 0; l < 3; ++l) {
        int kin = (l == 0) ? 80 : 512, kp = (l == 0) ? 96 : 512;
        for (int d = 0; d < 2; ++d) {
            k_cvt_pad<<<CG, 256, 0, stream>>>(Wih[l][d], WihF[l][d], 1024, kin, kp);
            k_cvt_pad<<<CG, 256, 0, stream>>>(Whh[l][d], WhhF[l][d], 1024, 256, 256);
        }
    }
    k_cvt_pad<<<CG, 256, 0, stream>>>(W1, W1F, 256, 512, 512);
    k_cvt_pad<<<CG, 256, 0, stream>>>(W2, W2F, 256, 256, 256);
    k_make_decW<<<CG, 256, 0, stream>>>(dWih, dWhh, WdF);
    k_make_tokW<<<128, 256, 0, stream>>>(dWih, tokW);

    // encoder: per layer, precompute xz (both dirs) then scan
    const _Float16* in = xpad; int lda = 96, kit = 3;
    _Float16* bufs[2] = {actA, actB};
    for (int l = 0; l < 3; ++l) {
        // N=1024, NT=4: 8 waves x 64 cols = 512 cols/block -> grid.y = 2
        k_gemm16<4><<<dim3(ROWS / 16, 2), 256, 0, stream>>>(in, lda, WihF[l][0], lda, bE[l][0], xzf, 1024, kit);
        k_gemm16<4><<<dim3(ROWS / 16, 2), 256, 0, stream>>>(in, lda, WihF[l][1], lda, bE[l][1], xzb, 1024, kit);
        _Float16* ob = bufs[l & 1];
        k_lstm_scan<<<2, 1024, 0, stream>>>(xzf, xzb, WhhF[l][0], WhhF[l][1], ob);
        in = ob; lda = 512; kit = 16;
    }
    // enc_proj = enc @ W1^T ; N=256, NT=2: 8 waves x 32 cols = 256 -> grid.y = 1
    k_gemm16<2><<<dim3(ROWS / 16, 1), 256, 0, stream>>>(in, 512, W1F, 512, nullptr, eproj, 256, 16);

    // decoder
    size_t smem = (size_t)B_ * T_ * 4 + (size_t)B_ * H_ * 4 + (size_t)H_ * 4 + (size_t)B_ * 768 * 2;
    k_decoder<<<1, 1024, smem, stream>>>(eproj, in, W2F, WdF, tokW, dec_b, V, Wout, bout,
                                         targets, (float*)d_out);
}
